// MotionGRU_20272245637222
// MI455X (gfx1250) — matrix operand order repaired and checked by reading, not compiled, not run on hardware
//
#include <hip/hip_runtime.h>
#include <hip/hip_bf16.h>
#include <math.h>

typedef __attribute__((ext_vector_type(16))) __bf16 v16bf;
typedef __attribute__((ext_vector_type(8)))  float  v8f;
typedef unsigned int   u32;
typedef unsigned short u16;

constexpr int Bn  = 16;
constexpr int HC  = 256;
constexpr int C4  = 64;
constexpr int Hh  = 112, Ww = 112;
constexpr int H2  = 56,  W2 = 56;
constexpr int HW  = Hh * Ww;     // 12544
constexpr int HW2 = H2 * W2;     // 3136
constexpr int K2  = 9;
constexpr int NPIX2 = Bn * HW2;  // 50176
constexpr int NPIX  = Bn * HW;   // 200704

__device__ __forceinline__ u16 f2bfbits(float f) {
    u32 u = __builtin_bit_cast(u32, f);
    u32 r = u + 0x7FFFu + ((u >> 16) & 1u);   // round-to-nearest-even
    return (u16)(r >> 16);
}
__device__ __forceinline__ u32 pack2bf(float a, float b) {
    return (u32)f2bfbits(a) | ((u32)f2bfbits(b) << 16);
}
__device__ __forceinline__ float sigmoidf(float x) { return 1.f / (1.f + __expf(-x)); }
__device__ __forceinline__ int clampi(int v, int lo, int hi) { return v < lo ? lo : (v > hi ? hi : v); }

// ---- WMMA fragment layouts used here (contiguous k, not the ISA 7.12.2 table order) ----
// A 16x32 bf16: lane L holds M=L%16, lane-half h; element e <-> K = 16h+e (contiguous, matches B).
// We store A tiles in LDS *in fragment order*: u16 Abuf[32 lanes][16 elems] per 16x32 chunk.
// B 32x16 bf16: lane L holds N=L%16, half kh=L/16, elements j=0..15 <-> K = k0+16*kh+j
// (contiguous in K!). Weights are pre-converted to this order in global memory.

// position of (m, kk) inside a 512-u16 A-fragment chunk (kk in [0,32))
__device__ __forceinline__ int afrag_idx(int m, int kk) {
    int hl = kk >> 4;
    return (m + 16 * hl) * 16 + (kk & 15);
}
// u32 index for an even-k pair (kk even)
__device__ __forceinline__ int afrag_pair_idx(int m, int kk) {
    int hl = kk >> 4;
    return (m + 16 * hl) * 8 + ((kk & 15) >> 1);
}
__device__ __forceinline__ v16bf load_frag_lds(const u16* buf, int lane) {
    return *reinterpret_cast<const v16bf*>(buf + lane * 16);
}
__device__ __forceinline__ v16bf load_frag_g(const u16* __restrict__ buf, long blk, int lane) {
    return *reinterpret_cast<const v16bf*>(buf + (blk * 32 + lane) * 16);
}

// ================= Weight pre-conversion (fp32 -> bf16, B-fragment order) =================
// enc: K ordered as k = tap*256 + ci ; source enc_w[n][ci][tap] = enc_w[(n*256+ci)*9+tap]
__global__ __launch_bounds__(256) void conv_w_enc_kernel(const float* __restrict__ enc_w,
                                                         u16* __restrict__ out) {
    int gid = blockIdx.x * 256 + threadIdx.x;        // one thread per (nt,kc,lane)
    if (gid >= 4 * 72 * 32) return;
    int lane = gid & 31;
    int kc   = (gid >> 5) % 72;
    int nt   = (gid >> 5) / 72;
    int n    = nt * 16 + (lane & 15);
    int kbase = kc * 32 + 16 * (lane >> 4);
    u16* dst = out + (long)gid * 16;
#pragma unroll
    for (int j = 0; j < 16; ++j) {
        int k = kbase + j;
        int tap = k >> 8, ci = k & 255;
        dst[j] = f2bfbits(enc_w[(n * 256 + ci) * 9 + tap]);
    }
}
// flat [n][k] weights (dec1: ldw=576, out: ldw=512)
__global__ __launch_bounds__(256) void conv_w_flat_kernel(const float* __restrict__ w,
                                                          u16* __restrict__ out,
                                                          int NT, int NKC, int ldw) {
    int gid = blockIdx.x * 256 + threadIdx.x;
    if (gid >= NT * NKC * 32) return;
    int lane = gid & 31;
    int kc   = (gid >> 5) % NKC;
    int nt   = (gid >> 5) / NKC;
    int n    = nt * 16 + (lane & 15);
    int kbase = kc * 32 + 16 * (lane >> 4);
    const float* src = w + (long)n * ldw + kbase;
    u16* dst = out + (long)gid * 16;
#pragma unroll
    for (int j = 0; j < 16; ++j) dst[j] = f2bfbits(src[j]);
}

// ================= Kernel 1: enc conv 3x3 s2 (256->64) implicit GEMM =================
// Tile: 16 pixels x 64 out-ch (4 waves), K = 2304 ordered k = tap*256+ci, chunks of 128.
__global__ __launch_bounds__(128) void enc_conv_kernel(
    const float* __restrict__ h, const u16* __restrict__ encBf,
    const float* __restrict__ enc_b, float* __restrict__ enc_h)
{
    __shared__ __align__(32) u16 Abuf[4 * 512];      // 4 sub-chunks of 16x32 in fragment order
    const int tid = threadIdx.x, lane = tid & 31, wave = tid >> 5;
    const int pix0 = blockIdx.x * 16;
    // fixed pixel per thread for staging: 16 px x 8 k-slots
    const int spx = tid >> 3, skl = tid & 7;
    const int spidx = pix0 + spx;
    const int sb = spidx / HW2, srem = spidx % HW2;
    const int soy = srem / W2, sox = srem % W2;
    const float* hb = h + (long)sb * HC * HW;
    v8f acc = {};
    for (int c = 0; c < 18; ++c) {                   // 18 chunks of K=128
        const int k0 = c * 128;
#pragma unroll
        for (int q = 0; q < 8; ++q) {
            int kev = (skl + 8 * q) << 1;            // even offset in [0,128)
            int k = k0 + kev;
            int tap = k >> 8, ci = k & 255;          // shifts, no division
            int dy = tap / 3, dx = tap % 3;
            int iy = 2 * soy - 1 + dy, ix = 2 * sox - 1 + dx;
            float v0 = 0.f, v1 = 0.f;
            if (iy >= 0 && iy < Hh && ix >= 0 && ix < Ww) {
                const float* p = hb + ((long)ci * Hh + iy) * Ww + ix;
                v0 = p[0];
                v1 = p[(long)HW];                    // ci+1, same (iy,ix)
            }
            ((u32*)Abuf)[(kev >> 5) * 256 + afrag_pair_idx(spx, kev & 31)] = pack2bf(v0, v1);
        }
        __syncthreads();
#pragma unroll
        for (int c2 = 0; c2 < 4; ++c2) {
            v16bf a  = load_frag_lds(&Abuf[c2 * 512], lane);
            v16bf bf = load_frag_g(encBf, (long)wave * 72 + (c * 4 + c2), lane);
            acc = __builtin_amdgcn_wmma_f32_16x16x32_bf16(false, a, false, bf, (short)0, acc, false, false);
        }
        __syncthreads();
    }
    const int n = lane & 15, hm = lane >> 4;
    const int co = wave * 16 + n;
    const float bias = enc_b[co];
#pragma unroll
    for (int r = 0; r < 8; ++r) {
        int m = r + 8 * hm;
        int pidx = pix0 + m;
        int b = pidx / HW2, rem = pidx % HW2;
        enc_h[((long)b * C4 + co) * HW2 + rem] = acc[r] + bias;
    }
}

// ================= Kernel 2: GRU pointwise (u,r,z 1x1 convs + state update) =================
__global__ __launch_bounds__(128) void gru_pointwise_kernel(
    const float* __restrict__ enc_h, const float* __restrict__ f, const float* __restrict__ d,
    const float* __restrict__ u_w, const float* __restrict__ u_b,
    const float* __restrict__ r_w, const float* __restrict__ r_b,
    const float* __restrict__ z_w, const float* __restrict__ z_b,
    float* __restrict__ f_new, float* __restrict__ d_new)
{
    __shared__ float Wu[18 * 82], Wr[18 * 82], Wz[18 * 82];
    __shared__ float Bu[18], Br[18], Bz[18];
    const int tid = threadIdx.x;
    for (int i = tid; i < 18 * 82; i += 128) { Wu[i] = u_w[i]; Wr[i] = r_w[i]; Wz[i] = z_w[i]; }
    for (int i = tid; i < 18; i += 128)      { Bu[i] = u_b[i]; Br[i] = r_b[i]; Bz[i] = z_b[i]; }
    __syncthreads();
    const int pidx = blockIdx.x * 128 + tid;
    if (pidx >= NPIX2) return;
    const int b = pidx / HW2, rem = pidx % HW2;
    float ench[64], fv[18], dv[18], uv[18], rv[18];
    for (int c = 0; c < 64; ++c) ench[c] = enc_h[((long)b * C4 + c) * HW2 + rem];
    for (int c = 0; c < 18; ++c) fv[c] = f[((long)b * 18 + c) * HW2 + rem];
    for (int c = 0; c < 18; ++c) dv[c] = d[((long)b * 18 + c) * HW2 + rem];
    for (int o = 0; o < 18; ++o) {
        float su = Bu[o], sr = Br[o];
        const float* wu = &Wu[o * 82]; const float* wr = &Wr[o * 82];
        for (int c = 0; c < 64; ++c) { su += wu[c] * ench[c]; sr += wr[c] * ench[c]; }
        for (int c = 0; c < 18; ++c) { su += wu[64 + c] * fv[c]; sr += wr[64 + c] * fv[c]; }
        uv[o] = sigmoidf(su);
        rv[o] = sigmoidf(sr);
    }
    for (int o = 0; o < 18; ++o) {
        float sz = Bz[o];
        const float* wz = &Wz[o * 82];
        for (int c = 0; c < 64; ++c) sz += wz[c] * ench[c];
        for (int c = 0; c < 18; ++c) sz += wz[64 + c] * (rv[c] * fv[c]);
        float zv = tanhf(sz);
        float fu = uv[o] * zv + (1.f - uv[o]) * fv[o];
        float dn = dv[o] + 0.5f * (fv[o] - dv[o]);   // ALPHA = 0.5
        long off = ((long)b * 18 + o) * HW2 + rem;
        f_new[off] = fu + dn;
        d_new[off] = dn;
    }
}

// ================= Kernel 3: m = sigmoid(conv3x3(enc_h, 64->9)) =================
__global__ __launch_bounds__(128) void m_conv_kernel(
    const float* __restrict__ enc_h, const float* __restrict__ m_w,
    const float* __restrict__ m_b, float* __restrict__ m_out)
{
    __shared__ float Wm[9 * 64 * 9];
    const int tid = threadIdx.x;
    for (int i = tid; i < 9 * 64 * 9; i += 128) Wm[i] = m_w[i];
    __syncthreads();
    const int pidx = blockIdx.x * 128 + tid;
    if (pidx >= NPIX2) return;
    const int b = pidx / HW2, rem = pidx % HW2;
    const int oy = rem / W2, ox = rem % W2;
    float acc[9];
    for (int o = 0; o < 9; ++o) acc[o] = m_b[o];
    for (int ci = 0; ci < 64; ++ci) {
        const float* base = enc_h + ((long)b * C4 + ci) * HW2;
        float nb[9];
#pragma unroll
        for (int t = 0; t < 9; ++t) {
            int iy = oy + t / 3 - 1, ix = ox + t % 3 - 1;
            nb[t] = (iy >= 0 && iy < H2 && ix >= 0 && ix < W2) ? base[iy * W2 + ix] : 0.f;
        }
#pragma unroll
        for (int o = 0; o < 9; ++o) {
            const float* w = &Wm[(o * 64 + ci) * 9];
            float s = 0.f;
#pragma unroll
            for (int t = 0; t < 9; ++t) s += w[t] * nb[t];
            acc[o] += s;
        }
    }
    for (int o = 0; o < 9; ++o)
        m_out[((long)b * 9 + o) * HW2 + rem] = sigmoidf(acc[o]);
}

// ======= Kernel 4: fused warp (9-offset bilinear * m) + dec1 1x1 (576->64) WMMA =======
__global__ __launch_bounds__(128) void warp_dec1_kernel(
    const float* __restrict__ enc_h, const float* __restrict__ f_new,
    const float* __restrict__ m_in, const u16* __restrict__ dec1Bf,
    const float* __restrict__ dec1_b, float* __restrict__ temp)
{
    __shared__ __align__(32) u16 Abuf[18 * 512];     // A[16][576] in fragment order, bf16
    const int tid = threadIdx.x, lane = tid & 31, wave = tid >> 5;
    const int pix0 = blockIdx.x * 16;
    for (int j = tid; j < 16 * 9; j += 128) {
        int px = j / 9, i = j % 9;
        int pidx = pix0 + px;
        int b = pidx / HW2, rem = pidx % HW2;
        int oy = rem / W2, ox = rem % W2;
        float fx = f_new[((long)b * 18 + i) * HW2 + rem];
        float fy = f_new[((long)b * 18 + 9 + i) * HW2 + rem];
        float pxf = (float)ox + (float)(i / 3 - 1) + fx;   // p_ix = i//3 - 1
        float pyf = (float)oy + (float)(i % 3 - 1) + fy;   // p_iy = i%3 - 1
        float x0f = floorf(pxf), y0f = floorf(pyf);
        int x0 = (int)x0f, y0 = (int)y0f;
        float wx1 = pxf - x0f, wx0 = 1.f - wx1;
        float wy1 = pyf - y0f, wy0 = 1.f - wy1;
        float vx0 = (x0 >= 0 && x0 < W2) ? 1.f : 0.f;
        float vx1 = (x0 + 1 >= 0 && x0 + 1 < W2) ? 1.f : 0.f;
        float vy0 = (y0 >= 0 && y0 < H2) ? 1.f : 0.f;
        float vy1 = (y0 + 1 >= 0 && y0 + 1 < H2) ? 1.f : 0.f;
        int x0c = clampi(x0, 0, W2 - 1), x1c = clampi(x0 + 1, 0, W2 - 1);
        int y0c = clampi(y0, 0, H2 - 1), y1c = clampi(y0 + 1, 0, H2 - 1);
        float w00 = wx0 * wy0 * vx0 * vy0, w01 = wx1 * wy0 * vx1 * vy0;
        float w10 = wx0 * wy1 * vx0 * vy1, w11 = wx1 * wy1 * vx1 * vy1;
        int i00 = y0c * W2 + x0c, i01 = y0c * W2 + x1c;
        int i10 = y1c * W2 + x0c, i11 = y1c * W2 + x1c;
        float mval = m_in[((long)b * 9 + i) * HW2 + rem];
        const float* base = enc_h + (long)b * C4 * HW2;
        for (int ci = 0; ci < C4; ++ci) {
            const float* ch = base + (long)ci * HW2;
            float v = w00 * ch[i00] + w01 * ch[i01] + w10 * ch[i10] + w11 * ch[i11];
            int k = ci * 9 + i;
            Abuf[(k >> 5) * 512 + afrag_idx(px, k & 31)] = f2bfbits(mval * v);
        }
    }
    __syncthreads();
    v8f acc = {};
    for (int kc = 0; kc < 18; ++kc) {                // K = 576
        v16bf a  = load_frag_lds(&Abuf[kc * 512], lane);
        v16bf bf = load_frag_g(dec1Bf, (long)wave * 18 + kc, lane);
        acc = __builtin_amdgcn_wmma_f32_16x16x32_bf16(false, a, false, bf, (short)0, acc, false, false);
    }
    const int n = lane & 15, hm = lane >> 4;
    const int co = wave * 16 + n;
    const float bias = dec1_b[co];
#pragma unroll
    for (int r = 0; r < 8; ++r) {
        int m = r + 8 * hm;
        int pidx = pix0 + m;
        int b = pidx / HW2, rem = pidx % HW2;
        temp[((long)b * C4 + co) * HW2 + rem] = acc[r] + bias;
    }
}

// ======= Kernel 5: fused deconv(64->256,s2) + out 1x1 (512->256) WMMA + gate =======
// dec_h tile lives only in LDS; fp32 copy kept for exact gating, bf16 fragment copy for GEMM.
__global__ __launch_bounds__(512) void deconv_out_gate_kernel(
    const float* __restrict__ h, const float* __restrict__ temp,
    const float* __restrict__ deconv_w, const float* __restrict__ deconv_b,
    const u16* __restrict__ outBf, const float* __restrict__ out_b,
    float* __restrict__ x_out)
{
    __shared__ float As[16 * 516];                   // [px]: [0,256)=dec_h, [256,512)=h (fp32)
    __shared__ __align__(32) u16 Abuf[16 * 512];     // same tile, bf16 fragment order
    const int tid = threadIdx.x, lane = tid & 31, wave = tid >> 5;
    const int pix0 = blockIdx.x * 16;
    for (int j = tid; j < 16 * 256; j += 512) {
        int px = j >> 8, co = j & 255;
        int pidx = pix0 + px;
        int b = pidx / HW, rem = pidx % HW;
        int oy = rem / Ww, ox = rem % Ww;
        float s = deconv_b[co];
        int kys[2], iys[2], nky, kxs[2], ixs[2], nkx;
        if ((oy & 1) == 0) { kys[0] = 1; iys[0] = oy >> 1; nky = 1; }
        else { kys[0] = 0; iys[0] = (oy + 1) >> 1; kys[1] = 2; iys[1] = (oy - 1) >> 1; nky = 2; }
        if ((ox & 1) == 0) { kxs[0] = 1; ixs[0] = ox >> 1; nkx = 1; }
        else { kxs[0] = 0; ixs[0] = (ox + 1) >> 1; kxs[1] = 2; ixs[1] = (ox - 1) >> 1; nkx = 2; }
        for (int yy = 0; yy < nky; ++yy) {
            int iy = iys[yy];
            if (iy < 0 || iy >= H2) continue;
            for (int xx = 0; xx < nkx; ++xx) {
                int ix = ixs[xx];
                if (ix < 0 || ix >= W2) continue;
                int tapw = kys[yy] * 3 + kxs[xx];
                const float* tp = temp + (long)b * C4 * HW2 + iy * W2 + ix;
                const float* wp = deconv_w + (long)co * 9 + tapw;   // + ci*HC*9
                for (int ci = 0; ci < C4; ++ci)
                    s += tp[(long)ci * HW2] * wp[(long)ci * HC * 9];
            }
        }
        float hv = h[((long)b * HC + co) * HW + rem];
        As[px * 516 + co]       = s;
        As[px * 516 + 256 + co] = hv;
        int cd = co, ch = 256 + co;
        Abuf[(cd >> 5) * 512 + afrag_idx(px, cd & 31)] = f2bfbits(s);
        Abuf[(ch >> 5) * 512 + afrag_idx(px, ch & 31)] = f2bfbits(hv);
    }
    __syncthreads();
    v8f acc = {};
    for (int kc = 0; kc < 16; ++kc) {                // K = 512
        v16bf a  = load_frag_lds(&Abuf[kc * 512], lane);
        v16bf bf = load_frag_g(outBf, (long)wave * 16 + kc, lane);
        acc = __builtin_amdgcn_wmma_f32_16x16x32_bf16(false, a, false, bf, (short)0, acc, false, false);
    }
    const int n = lane & 15, hm = lane >> 4;
    const int co = wave * 16 + n;
    const float bias = out_b[co];
#pragma unroll
    for (int r = 0; r < 8; ++r) {
        int m = r + 8 * hm;
        int pidx = pix0 + m;
        int b = pidx / HW, rem = pidx % HW;
        float g  = sigmoidf(acc[r] + bias);
        float hv = As[m * 516 + 256 + co];
        float dv = As[m * 516 + co];
        x_out[((long)b * HC + co) * HW + rem] = g * hv + (1.f - g) * dv;
    }
}

extern "C" void kernel_launch(void* const* d_in, const int* in_sizes, int n_in,
                              void* d_out, int out_size, void* d_ws, size_t ws_size,
                              hipStream_t stream) {
    const float* h        = (const float*)d_in[0];
    const float* f        = (const float*)d_in[1];
    const float* d        = (const float*)d_in[2];
    const float* enc_w    = (const float*)d_in[3];
    const float* enc_b    = (const float*)d_in[4];
    const float* u_w      = (const float*)d_in[5];
    const float* u_b      = (const float*)d_in[6];
    const float* r_w      = (const float*)d_in[7];
    const float* r_b      = (const float*)d_in[8];
    const float* z_w      = (const float*)d_in[9];
    const float* z_b      = (const float*)d_in[10];
    const float* m_w      = (const float*)d_in[11];
    const float* m_b      = (const float*)d_in[12];
    const float* dec1_w   = (const float*)d_in[13];
    const float* dec1_b   = (const float*)d_in[14];
    const float* deconv_w = (const float*)d_in[15];
    const float* deconv_b = (const float*)d_in[16];
    const float* out_w    = (const float*)d_in[17];
    const float* out_b    = (const float*)d_in[18];

    // outputs: x, f_new, d_new (concatenated)
    float* x_out = (float*)d_out;
    float* f_new = x_out + (long)Bn * HC * HW;
    float* d_new = f_new + (long)Bn * 18 * HW2;

    // workspace: enc_h (12.8MB) + m (1.8MB) + temp (12.8MB) + bf16 weight fragments (0.6MB)
    float* enc_h = (float*)d_ws;
    float* m_buf = enc_h + (long)Bn * C4 * HW2;
    float* temp  = m_buf + (long)Bn * K2 * HW2;
    u16*   encBf  = (u16*)(temp + (long)Bn * C4 * HW2);
    u16*   dec1Bf = encBf + (long)4 * 72 * 32 * 16;   // 147456
    u16*   outBf  = dec1Bf + (long)4 * 18 * 32 * 16;  // 36864

    conv_w_enc_kernel<<<(4 * 72 * 32 + 255) / 256, 256, 0, stream>>>(enc_w, encBf);
    conv_w_flat_kernel<<<(4 * 18 * 32 + 255) / 256, 256, 0, stream>>>(dec1_w, dec1Bf, 4, 18, 576);
    conv_w_flat_kernel<<<(16 * 16 * 32 + 255) / 256, 256, 0, stream>>>(out_w, outBf, 16, 16, 512);

    enc_conv_kernel<<<NPIX2 / 16, 128, 0, stream>>>(h, encBf, enc_b, enc_h);
    gru_pointwise_kernel<<<(NPIX2 + 127) / 128, 128, 0, stream>>>(
        enc_h, f, d, u_w, u_b, r_w, r_b, z_w, z_b, f_new, d_new);
    m_conv_kernel<<<(NPIX2 + 127) / 128, 128, 0, stream>>>(enc_h, m_w, m_b, m_buf);
    warp_dec1_kernel<<<NPIX2 / 16, 128, 0, stream>>>(enc_h, f_new, m_buf, dec1Bf, dec1_b, temp);
    deconv_out_gate_kernel<<<NPIX / 16, 512, 0, stream>>>(
        h, temp, deconv_w, deconv_b, outBf, out_b, x_out);
}
